// GCNModifiedAE2_72756745994508
// MI455X (gfx1250) — matrix operand order repaired and checked by reading, not compiled, not run on hardware
//
#include <hip/hip_runtime.h>
#include <hip/hip_bf16.h>

typedef __bf16 bf16_t;
typedef __attribute__((ext_vector_type(16))) __bf16 v16bf;
typedef __attribute__((ext_vector_type(8)))  float  v8f;
typedef __attribute__((ext_vector_type(4)))  int    v4i;

#define BM 64
#define BN 64
#define BK 32

// Async global->LDS staging path (gfx1250 GLOBAL_LOAD_ASYNC_TO_LDS_B128,
// tracked with ASYNCcnt). Gated so the file compiles on toolchains where the
// builtins are absent or differently named.  The builtin takes (int4* global,
// int4* lds, imm offset, imm cpol) — generic pointers convert implicitly to
// the __device__/__shared__ address-space-qualified parameter types.
#if defined(__gfx1250__) &&                                          \
    __has_builtin(__builtin_amdgcn_global_load_async_to_lds_b128) && \
    __has_builtin(__builtin_amdgcn_s_wait_asynccnt)
#define GEMM_USE_ASYNC 1
#else
#define GEMM_USE_ASYNC 0
#endif

// ---------------------------------------------------------------------------
// f32 -> bf16 conversion, optional ReLU. 4 elements / thread.
// ---------------------------------------------------------------------------
__global__ __launch_bounds__(256) void cvt_f32_bf16(const float* __restrict__ in,
                                                    bf16_t* __restrict__ out,
                                                    int n, int do_relu) {
  int i = (blockIdx.x * blockDim.x + threadIdx.x) * 4;
  if (i + 3 < n) {
    float4 v = *reinterpret_cast<const float4*>(in + i);
    if (do_relu) {
      v.x = fmaxf(v.x, 0.0f); v.y = fmaxf(v.y, 0.0f);
      v.z = fmaxf(v.z, 0.0f); v.w = fmaxf(v.w, 0.0f);
    }
    bf16_t o[4] = {(bf16_t)v.x, (bf16_t)v.y, (bf16_t)v.z, (bf16_t)v.w};
    *reinterpret_cast<uint2*>(out + i) = *reinterpret_cast<const uint2*>(o);
  }
}

// ---------------------------------------------------------------------------
// Edge scatter: agg[row] += w * support[col].  One wave per edge, 8 f32/lane,
// f32 atomics into L2 (agg = 16 MB, resident in the 192 MB L2).
// ---------------------------------------------------------------------------
__global__ __launch_bounds__(256) void edge_scatter(const int* __restrict__ erow,
                                                    const int* __restrict__ ecol,
                                                    const float* __restrict__ ew,
                                                    const float* __restrict__ support,
                                                    float* __restrict__ agg, int E) {
  int gtid = blockIdx.x * blockDim.x + threadIdx.x;
  int wave = gtid >> 5;
  int lane = gtid & 31;
  int nwaves = (gridDim.x * blockDim.x) >> 5;
  for (int e = wave; e < E; e += nwaves) {
    int r = erow[e];
    int c = ecol[e];
    float w = ew[e];
    const float4* src = reinterpret_cast<const float4*>(support + (size_t)c * 256);
    float4 a = src[lane * 2 + 0];
    float4 b = src[lane * 2 + 1];
    float* dst = agg + (size_t)r * 256 + lane * 8;
    __hip_atomic_fetch_add(dst + 0, w * a.x, __ATOMIC_RELAXED, __HIP_MEMORY_SCOPE_AGENT);
    __hip_atomic_fetch_add(dst + 1, w * a.y, __ATOMIC_RELAXED, __HIP_MEMORY_SCOPE_AGENT);
    __hip_atomic_fetch_add(dst + 2, w * a.z, __ATOMIC_RELAXED, __HIP_MEMORY_SCOPE_AGENT);
    __hip_atomic_fetch_add(dst + 3, w * a.w, __ATOMIC_RELAXED, __HIP_MEMORY_SCOPE_AGENT);
    __hip_atomic_fetch_add(dst + 4, w * b.x, __ATOMIC_RELAXED, __HIP_MEMORY_SCOPE_AGENT);
    __hip_atomic_fetch_add(dst + 5, w * b.y, __ATOMIC_RELAXED, __HIP_MEMORY_SCOPE_AGENT);
    __hip_atomic_fetch_add(dst + 6, w * b.z, __ATOMIC_RELAXED, __HIP_MEMORY_SCOPE_AGENT);
    __hip_atomic_fetch_add(dst + 7, w * b.w, __ATOMIC_RELAXED, __HIP_MEMORY_SCOPE_AGENT);
  }
}

// ---------------------------------------------------------------------------
// bf16 WMMA GEMM, f32 accumulate.  C[m,n] = sum_k A[m,k] * Bop[k,n] (+bias[n])
//   TRANS_B=0 : Bop[k,n] = B[k*ldb + n]       (row-major KxN)
//   TRANS_B=1 : Bop[k,n] = B[n*ldb + k]       (row-major NxK, i.e. @ W.T / @ X.T)
// Block = 128 threads (4 waves), 64x64 tile; wave = 2x2 grid of 16x16 WMMA
// tiles; K stepped by 32 through LDS.  Wave32 VGPR layouts per ISA 7.12.2.
// Tiles staged with GLOBAL_LOAD_ASYNC_TO_LDS_B128 when available.
// M,N must be multiples of 64; K multiple of 32 (true for all calls here).
// ---------------------------------------------------------------------------
template <bool TRANS_B>
__global__ __launch_bounds__(128) void gemm_bf16(const bf16_t* __restrict__ A,
                                                 const bf16_t* __restrict__ B,
                                                 float* __restrict__ C,
                                                 const float* __restrict__ bias,
                                                 int M, int N, int K,
                                                 int lda, int ldb, int ldc) {
  __shared__ bf16_t As[BM][BK];   // row-major MxK chunk
  __shared__ bf16_t Bs[BN][BK];   // K-transposed: Bs[n][k] = Bop[k][n]

  const int tid  = threadIdx.x;        // 0..127
  const int lane = tid & 31;
  const int wave = tid >> 5;           // 0..3
  const int wm   = (wave >> 1) * 32;   // wave row offset inside 64x64 tile
  const int wn   = (wave & 1)  * 32;   // wave col offset
  const int m0   = blockIdx.y * BM;
  const int n0   = blockIdx.x * BN;
  const int lhalf = lane >> 4;         // 0: lanes 0-15, 1: lanes 16-31
  const int l16   = lane & 15;

  v8f acc[2][2] = {};                  // zero-initialized accumulators

  for (int k0 = 0; k0 < K; k0 += BK) {
    // --- stage A tile: 64 rows x 32 bf16; each thread moves 8 bf16 (16 B)
    {
      int r  = tid >> 1;
      int ck = (tid & 1) * 16;
      const bf16_t* src = A + (size_t)(m0 + r) * lda + (k0 + ck);
#if GEMM_USE_ASYNC
      __builtin_amdgcn_global_load_async_to_lds_b128((v4i*)src, (v4i*)&As[r][ck], 0, 0);
#else
      *reinterpret_cast<uint4*>(&As[r][ck]) = *reinterpret_cast<const uint4*>(src);
#endif
      if (k0 + BK < K) __builtin_prefetch(src + BK, 0, 1);   // global_prefetch_b8
    }
    // --- stage B tile into Bs[n][k]
    if (TRANS_B) {
      int r  = tid >> 1;                 // n within tile
      int ck = (tid & 1) * 16;
      const bf16_t* src = B + (size_t)(n0 + r) * ldb + (k0 + ck);
#if GEMM_USE_ASYNC
      __builtin_amdgcn_global_load_async_to_lds_b128((v4i*)src, (v4i*)&Bs[r][ck], 0, 0);
#else
      *reinterpret_cast<uint4*>(&Bs[r][ck]) = *reinterpret_cast<const uint4*>(src);
#endif
      if (k0 + BK < K) __builtin_prefetch(src + BK, 0, 1);
    } else {
      int n  = tid >> 1;
      int kk = (tid & 1) * 16;
      #pragma unroll
      for (int q = 0; q < 16; ++q)
        Bs[n][kk + q] = B[(size_t)(k0 + kk + q) * ldb + (n0 + n)];
    }
#if GEMM_USE_ASYNC
    __builtin_amdgcn_s_wait_asynccnt(0);   // async tiles landed in LDS
#endif
    __syncthreads();

    // --- build fragments (ISA 7.12.2 wave32 layouts)
    // A 16x32: lane holds row M=l16; elems[0..7]=K(lhalf*8+0..7),
    //                                elems[8..15]=K(16+lhalf*8+0..7)
    v16bf afrag[2];
    #pragma unroll
    for (int i = 0; i < 2; ++i) {
      int mrow = wm + i * 16 + l16;
      union { uint4 u[2]; v16bf v; } ua;
      ua.u[0] = *reinterpret_cast<const uint4*>(&As[mrow][lhalf * 8]);
      ua.u[1] = *reinterpret_cast<const uint4*>(&As[mrow][16 + lhalf * 8]);
      afrag[i] = ua.v;
    }
    // B 32x16: lane holds col N=l16; elems[0..7]=K(lhalf*8+0..7), elems[8..15]=K(16+lhalf*8+0..7)
    v16bf bfrag[2];
    #pragma unroll
    for (int j = 0; j < 2; ++j) {
      int ncol = wn + j * 16 + l16;
      union { uint4 u[2]; v16bf v; } ub;
      ub.u[0] = *reinterpret_cast<const uint4*>(&Bs[ncol][lhalf * 8]);
      ub.u[1] = *reinterpret_cast<const uint4*>(&Bs[ncol][16 + lhalf * 8]);
      bfrag[j] = ub.v;
    }

    #pragma unroll
    for (int i = 0; i < 2; ++i)
      #pragma unroll
      for (int j = 0; j < 2; ++j)
        acc[i][j] = __builtin_amdgcn_wmma_f32_16x16x32_bf16(
            false, afrag[i], false, bfrag[j], (short)0, acc[i][j], false, false);

    __syncthreads();
  }

  // --- store: D layout: lane holds col N=l16; elem r -> row (8*lhalf + r)
  #pragma unroll
  for (int i = 0; i < 2; ++i) {
    #pragma unroll
    for (int j = 0; j < 2; ++j) {
      int ncol  = n0 + wn + j * 16 + l16;
      float bv  = bias ? bias[ncol] : 0.0f;
      int mbase = m0 + wm + i * 16 + lhalf * 8;
      #pragma unroll
      for (int r = 0; r < 8; ++r)
        C[(size_t)(mbase + r) * ldc + ncol] = acc[i][j][r] + bv;
    }
  }
}

// ---------------------------------------------------------------------------
// Host-side launcher
// ---------------------------------------------------------------------------
extern "C" void kernel_launch(void* const* d_in, const int* in_sizes, int n_in,
                              void* d_out, int out_size, void* d_ws, size_t ws_size,
                              hipStream_t stream) {
  (void)n_in; (void)out_size; (void)ws_size;

  const float* x      = (const float*)d_in[0];   // [16384, 512]
  const int*   erow   = (const int*)  d_in[1];   // [E]
  const int*   ecol   = (const int*)  d_in[2];   // [E]
  const float* ew     = (const float*)d_in[3];   // [E]
  const float* gcW    = (const float*)d_in[4];   // [512, 256]
  const float* nodeW  = (const float*)d_in[5];   // [256, 256]
  const float* nodeB  = (const float*)d_in[6];   // [256]
  const float* neighW = (const float*)d_in[7];   // [256, 256]
  const float* neighB = (const float*)d_in[8];   // [256]

  const int Nn = 16384, F0 = 512, H = 256;
  const int E  = in_sizes[1];

  float* out = (float*)d_out;
  float* out_adj   = out;                                      // [16384,16384]
  float* out_node  = out + (size_t)Nn * Nn;                    // [16384,256]
  float* out_neigh = out + (size_t)Nn * Nn + (size_t)Nn * H;   // [16384,256]

  // workspace carve-up (~56.5 MB)
  char* ws = (char*)d_ws;
  bf16_t* xb      = (bf16_t*)(ws);                    // 16 MB : x as bf16
  bf16_t* gWb     = (bf16_t*)(ws + 16777216);         // 256 KB
  bf16_t* nWb     = (bf16_t*)(ws + 17039360);         // 128 KB
  bf16_t* ngWb    = (bf16_t*)(ws + 17170432);         // 128 KB
  float*  support = (float*) (ws + 17301504);         // 16 MB
  float*  agg     = (float*) (ws + 34078720);         // 16 MB
  bf16_t* H1b     = (bf16_t*)(ws + 50855936);         // 8 MB
  bf16_t* neb     = (bf16_t*)(ws);                    // reuse xb region
  bf16_t* nbb     = (bf16_t*)(ws + 8388608);          // xb region + 8 MB

  // 1) conversions to bf16
  cvt_f32_bf16<<<(Nn * F0 / 4 + 255) / 256, 256, 0, stream>>>(x, xb, Nn * F0, 0);
  cvt_f32_bf16<<<(F0 * H  / 4 + 255) / 256, 256, 0, stream>>>(gcW, gWb, F0 * H, 0);
  cvt_f32_bf16<<<(H * H   / 4 + 255) / 256, 256, 0, stream>>>(nodeW, nWb, H * H, 0);
  cvt_f32_bf16<<<(H * H   / 4 + 255) / 256, 256, 0, stream>>>(neighW, ngWb, H * H, 0);

  // 2) support = x @ gc_W   (B row-major KxN)
  gemm_bf16<false><<<dim3(H / BN, Nn / BM), 128, 0, stream>>>(
      xb, gWb, support, nullptr, Nn, H, F0, F0, H, H);

  // 3) agg = segment_sum(w * support[col] -> row)
  (void)hipMemsetAsync(agg, 0, (size_t)Nn * H * sizeof(float), stream);
  edge_scatter<<<4096, 256, 0, stream>>>(erow, ecol, ew, support, agg, E);

  // 4) H1 = relu(agg), as bf16
  cvt_f32_bf16<<<(Nn * H / 4 + 255) / 256, 256, 0, stream>>>(agg, H1b, Nn * H, 1);

  // 5) node_emb = H1 @ node_W.T + b ;  neighbor_emb = H1 @ neigh_W.T + b
  gemm_bf16<true><<<dim3(H / BN, Nn / BM), 128, 0, stream>>>(
      H1b, nWb, out_node, nodeB, Nn, H, H, H, H, H);
  gemm_bf16<true><<<dim3(H / BN, Nn / BM), 128, 0, stream>>>(
      H1b, ngWb, out_neigh, neighB, Nn, H, H, H, H, H);

  // 6) bf16 copies of the embeddings for the big GEMM
  cvt_f32_bf16<<<(Nn * H / 4 + 255) / 256, 256, 0, stream>>>(out_node, neb, Nn * H, 0);
  cvt_f32_bf16<<<(Nn * H / 4 + 255) / 256, 256, 0, stream>>>(out_neigh, nbb, Nn * H, 0);

  // 7) adj = node_emb @ neighbor_emb.T   (output-bandwidth-bound: 1 GB write)
  gemm_bf16<true><<<dim3(Nn / BN, Nn / BM), 128, 0, stream>>>(
      neb, nbb, out_adj, nullptr, Nn, Nn, H, H, H, Nn);
}
